// DGCNN12_23012434772602
// MI455X (gfx1250) — matrix operand order repaired and checked by reading, not compiled, not run on hardware
//
#include <hip/hip_runtime.h>

#define NGRAPH 64
#define NPG    1024          // nodes per graph
#define NNODE  (NGRAPH * NPG) // 65536
#define KNN    10
#define SLOPE  0.01f

typedef _Float16 v8h  __attribute__((ext_vector_type(8)));
typedef _Float16 v16h __attribute__((ext_vector_type(16)));
typedef float    v8f  __attribute__((ext_vector_type(8)));

__device__ __forceinline__ v16h cat8(v8h lo, v8h hi) {
  return __builtin_shufflevector(lo, hi, 0,1,2,3,4,5,6,7,8,9,10,11,12,13,14,15);
}
__device__ __forceinline__ float lrelu(float v) { return v >= 0.f ? v : SLOPE * v; }
__device__ __forceinline__ unsigned enc_ord(float f) {
  unsigned b = __float_as_uint(f);
  return (b & 0x80000000u) ? ~b : (b | 0x80000000u);
}
__device__ __forceinline__ float dec_ord(unsigned u) {
  return (u & 0x80000000u) ? __uint_as_float(u & 0x7fffffffu) : __uint_as_float(~u);
}

// ---------------- small prep kernels ----------------

__global__ __launch_bounds__(256) void k_init_pooled(unsigned* p, int n) {
  int i = blockIdx.x * 256 + threadIdx.x;
  if (i < n) p[i] = 0u;   // encoded minimum
}

// WT[c*R + r] = (f16) W[r*C + c]   (transpose K x N -> N x K for WMMA B loads)
__global__ __launch_bounds__(256) void k_transpose16(const float* __restrict__ W,
                                                     _Float16* __restrict__ WT,
                                                     int R, int C) {
  int e = blockIdx.x * 256 + threadIdx.x;
  if (e >= R * C) return;
  int r = e / C, c = e - r * C;
  WT[(size_t)c * R + r] = (_Float16)W[e];
}

// build xx f16 [NNODE][32] (5 features + zero pad) and sq norms (from f16 values)
__global__ __launch_bounds__(256) void k_build_xx(const float* __restrict__ x,
                                                  const float* __restrict__ pos,
                                                  const float* __restrict__ tq,
                                                  _Float16* __restrict__ xxf,
                                                  float* __restrict__ sq) {
  int i = blockIdx.x * 256 + threadIdx.x;
  if (i >= NNODE) return;
  _Float16 v[5];
  v[0] = (_Float16)tq[i];
  v[1] = (_Float16)x[i];
  v[2] = (_Float16)pos[3 * i + 0];
  v[3] = (_Float16)pos[3 * i + 1];
  v[4] = (_Float16)pos[3 * i + 2];
  _Float16* row = xxf + (size_t)i * 32;
  float s = 0.f;
#pragma unroll
  for (int f = 0; f < 5; ++f) { row[f] = v[f]; float fv = (float)v[f]; s += fv * fv; }
#pragma unroll
  for (int f = 5; f < 32; ++f) row[f] = (_Float16)0.f;
  sq[i] = s;
}

// sq norms of a 64-wide f16 slice of the hc slab (row stride 384)
__global__ __launch_bounds__(256) void k_sq64(const _Float16* __restrict__ h,
                                              float* __restrict__ sq) {
  int i = blockIdx.x * 256 + threadIdx.x;
  if (i >= NNODE) return;
  const _Float16* r = h + (size_t)i * 384;
  float s = 0.f;
#pragma unroll 8
  for (int f = 0; f < 64; ++f) { float v = (float)r[f]; s += v * v; }
  sq[i] = s;
}

// ---------------- fused Gram (WMMA) + kNN top-10 ----------------
// One wave per 16-row strip of one graph. KPAD = padded feature count (32 or 64).
template <int KPAD>
__global__ __launch_bounds__(32) void k_dist_topk(const _Float16* __restrict__ h,
                                                  int ld,
                                                  const float* __restrict__ sq,
                                                  int* __restrict__ idx) {
  __shared__ float ldsD[16][17];
  int wg   = blockIdx.x;        // 0 .. NGRAPH*64-1
  int g    = wg >> 6;
  int row0 = (wg & 63) << 4;
  int lane = threadIdx.x;
  int m    = lane & 15;
  int kb   = (lane < 16) ? 0 : 16;   // A-matrix K sub-block
  int kbB  = (lane < 16) ? 0 : 16;   // B-matrix K sub-block
  int nB   = lane & 15;
  int moff = (lane < 16) ? 0 : 8;

  const _Float16* hg  = h + (size_t)g * NPG * ld;
  const float*    sqg = sq + (size_t)g * NPG;

  // A tiles (rows of this strip), contiguous k = 16h + s (matches B)
  v16h a0, a1;
  {
    const _Float16* rp = hg + (size_t)(row0 + m) * ld;
    a0 = cat8(*(const v8h*)(rp + kb), *(const v8h*)(rp + 8 + kb));
    if (KPAD == 64)
      a1 = cat8(*(const v8h*)(rp + 32 + kb), *(const v8h*)(rp + 40 + kb));
  }
  float sqRow[8];
#pragma unroll
  for (int r = 0; r < 8; ++r) sqRow[r] = sqg[row0 + r + moff];

  float dk[KNN]; int ik[KNN];
#pragma unroll
  for (int t = 0; t < KNN; ++t) { dk[t] = __builtin_inff(); ik[t] = 0; }

  for (int ct = 0; ct < 64; ++ct) {
    int col0 = ct << 4;
    v8f c = {0.f, 0.f, 0.f, 0.f, 0.f, 0.f, 0.f, 0.f};
    const _Float16* cp = hg + (size_t)(col0 + nB) * ld;
    v16h b0 = *(const v16h*)(cp + kbB);
    c = __builtin_amdgcn_wmma_f32_16x16x32_f16(false, a0, false, b0, (short)0, c, false, false);
    if (KPAD == 64) {
      v16h b1 = *(const v16h*)(cp + 32 + kbB);
      c = __builtin_amdgcn_wmma_f32_16x16x32_f16(false, a1, false, b1, (short)0, c, false, false);
    }
    float sqc = sqg[col0 + nB];
#pragma unroll
    for (int r = 0; r < 8; ++r)
      ldsD[r + moff][nB] = sqRow[r] + sqc - 2.0f * c[r];
    __syncthreads();
    if (lane < 16) {
#pragma unroll 4
      for (int n = 0; n < 16; ++n) {
        float d = ldsD[lane][n];
        int   j = col0 + n;
        if (d < dk[KNN - 1]) {
          dk[KNN - 1] = d; ik[KNN - 1] = j;
#pragma unroll
          for (int t = KNN - 1; t > 0; --t) {
            if (dk[t] < dk[t - 1]) {
              float td = dk[t]; dk[t] = dk[t - 1]; dk[t - 1] = td;
              int   ti = ik[t]; ik[t] = ik[t - 1]; ik[t - 1] = ti;
            }
          }
        }
      }
    }
    __syncthreads();
  }
  if (lane < 16) {
    int base = (g * NPG + row0 + lane) * KNN;
#pragma unroll
    for (int t = 0; t < KNN; ++t) idx[base + t] = g * NPG + ik[t];
  }
}

// ---------------- P/Q projections ----------------
// conv1: F=5, W1 is 10x64.  P = xx*(Wtop-Wbot)+b, Q = xx*Wbot
__global__ __launch_bounds__(256) void k_pq5(const _Float16* __restrict__ xxf,
                                             const float* __restrict__ W1,
                                             const float* __restrict__ b1,
                                             float* __restrict__ P,
                                             float* __restrict__ Q) {
  int e = blockIdx.x * 256 + threadIdx.x;   // e < NNODE*64
  int i = e >> 6, o = e & 63;
  const _Float16* row = xxf + (size_t)i * 32;
  float p = b1[o], q = 0.f;
#pragma unroll
  for (int f = 0; f < 5; ++f) {
    float hv = (float)row[f];
    float wt = W1[f * 64 + o];
    float wb = W1[(5 + f) * 64 + o];
    p += hv * (wt - wb);
    q += hv * wb;
  }
  P[e] = p; Q[e] = q;
}

// convs 2..6: F=64, W2 is 128x64; h slice in hc slab (row stride 384)
__global__ __launch_bounds__(256) void k_pq64(const _Float16* __restrict__ h,
                                              const float* __restrict__ W2,
                                              const float* __restrict__ b2,
                                              float* __restrict__ P,
                                              float* __restrict__ Q) {
  __shared__ float hs[4][64];
  int node0 = blockIdx.x * 4;
  int li = threadIdx.x;
  int ni = li >> 6, o = li & 63;
  hs[ni][o] = (float)h[(size_t)(node0 + ni) * 384 + o];
  __syncthreads();
  float p = b2[o], q = 0.f;
#pragma unroll 8
  for (int f = 0; f < 64; ++f) {
    float hv = hs[ni][f];
    float wt = W2[f * 64 + o];
    float wb = W2[(64 + f) * 64 + o];
    p += hv * (wt - wb);
    q += hv * wb;
  }
  size_t e = (size_t)(node0 + ni) * 64 + o;
  P[e] = p; Q[e] = q;
}

// h_out[i,o] = sum_k lrelu(P[i,o] + Q[knn_k(i),o]) -> hc slab (f16)
__global__ __launch_bounds__(256) void k_agg(const float* __restrict__ P,
                                             const float* __restrict__ Q,
                                             const int* __restrict__ idx,
                                             _Float16* __restrict__ hout) {
  int e = blockIdx.x * 256 + threadIdx.x;   // e < NNODE*64
  int i = e >> 6, o = e & 63;
  float p = P[e];
  const int* row = idx + (size_t)i * KNN;
  float acc = 0.f;
#pragma unroll
  for (int t = 0; t < KNN; ++t) {
    float v = p + Q[(size_t)row[t] * 64 + o];
    acc += lrelu(v);
  }
  hout[(size_t)i * 384 + o] = (_Float16)acc;
}

// ---------------- big WMMA GEMMs (register-blocked: 16 rows x 64 cols / wave) ----
// t1 = f16( lrelu(hc @ Wl1a + bl1a) )   M=NNODE K=384 N=512 ; B pre-transposed [512][384]
__global__ __launch_bounds__(128) void k_gemm1(const _Float16* __restrict__ hc,
                                               const _Float16* __restrict__ WaT,
                                               const float* __restrict__ bl1a,
                                               _Float16* __restrict__ t1) {
  int tile = blockIdx.x * 4 + (threadIdx.x >> 5);  // 4096 * 8 = 32768 waves
  int mt = tile >> 3;        // M tile (16 rows)
  int ng = tile & 7;         // N group (64 cols)
  int row0 = mt << 4, n0 = ng << 6;
  int lane = threadIdx.x & 31;
  int m = lane & 15, kb = (lane < 16) ? 0 : 16;
  int nB = lane & 15, kbB = (lane < 16) ? 0 : 16;
  int moff = (lane < 16) ? 0 : 8;
  const _Float16* arow = hc  + (size_t)(row0 + m) * 384;
  const _Float16* brow = WaT + (size_t)(n0 + nB) * 384;
  v8f c[4];
#pragma unroll
  for (int t = 0; t < 4; ++t) c[t] = (v8f){0.f, 0.f, 0.f, 0.f, 0.f, 0.f, 0.f, 0.f};
#pragma unroll
  for (int k0 = 0; k0 < 384; k0 += 32) {
    v16h a = cat8(*(const v8h*)(arow + k0 + kb), *(const v8h*)(arow + k0 + 8 + kb));
#pragma unroll
    for (int t = 0; t < 4; ++t) {
      v16h b = *(const v16h*)(brow + (size_t)(t * 16) * 384 + k0 + kbB);
      c[t] = __builtin_amdgcn_wmma_f32_16x16x32_f16(false, a, false, b, (short)0, c[t], false, false);
    }
  }
#pragma unroll
  for (int t = 0; t < 4; ++t) {
    float bias = bl1a[n0 + t * 16 + nB];
#pragma unroll
    for (int r = 0; r < 8; ++r) {
      float v = lrelu(c[t][r] + bias);
      t1[(size_t)(row0 + r + moff) * 512 + n0 + t * 16 + nB] = (_Float16)v;
    }
  }
}

// out0 = t1 @ Wl1b + bl1b, fused per-graph max-pool via monotone-uint atomicMax
__global__ __launch_bounds__(128) void k_gemm2_pool(const _Float16* __restrict__ t1,
                                                    const _Float16* __restrict__ WbT,
                                                    const float* __restrict__ bl1b,
                                                    unsigned* __restrict__ pooled) {
  int tile = blockIdx.x * 4 + (threadIdx.x >> 5);  // 32768 waves
  int mt = tile >> 3;
  int ng = tile & 7;
  int row0 = mt << 4, n0 = ng << 6;
  int lane = threadIdx.x & 31;
  int m = lane & 15, kb = (lane < 16) ? 0 : 16;
  int nB = lane & 15, kbB = (lane < 16) ? 0 : 16;
  const _Float16* arow = t1  + (size_t)(row0 + m) * 512;
  const _Float16* brow = WbT + (size_t)(n0 + nB) * 512;
  v8f c[4];
#pragma unroll
  for (int t = 0; t < 4; ++t) c[t] = (v8f){0.f, 0.f, 0.f, 0.f, 0.f, 0.f, 0.f, 0.f};
#pragma unroll
  for (int k0 = 0; k0 < 512; k0 += 32) {
    v16h a = cat8(*(const v8h*)(arow + k0 + kb), *(const v8h*)(arow + k0 + 8 + kb));
#pragma unroll
    for (int t = 0; t < 4; ++t) {
      v16h b = *(const v16h*)(brow + (size_t)(t * 16) * 512 + k0 + kbB);
      c[t] = __builtin_amdgcn_wmma_f32_16x16x32_f16(false, a, false, b, (short)0, c[t], false, false);
    }
  }
  int g = row0 >> 10;   // 16-row tiles never cross graph boundaries (1024 % 16 == 0)
#pragma unroll
  for (int t = 0; t < 4; ++t) {
    float bias = bl1b[n0 + t * 16 + nB];
    float vmax = c[t][0] + bias;
#pragma unroll
    for (int r = 1; r < 8; ++r) { float v = c[t][r] + bias; vmax = v > vmax ? v : vmax; }
    atomicMax(&pooled[g * 512 + n0 + t * 16 + nB], enc_ord(vmax));
  }
}

// ---------------- head: pooled -> lrelu(@Wm1) -> @Wm2 ----------------
__global__ __launch_bounds__(256) void k_head(const unsigned* __restrict__ pooled,
                                              const float* __restrict__ Wm1,
                                              const float* __restrict__ bm1,
                                              const float* __restrict__ Wm2,
                                              const float* __restrict__ bm2,
                                              float* __restrict__ out) {
  __shared__ float pool[512];
  __shared__ float t2[256];
  int g = blockIdx.x, tid = threadIdx.x;
  pool[tid]       = dec_ord(pooled[g * 512 + tid]);
  pool[tid + 256] = dec_ord(pooled[g * 512 + tid + 256]);
  __syncthreads();
  float acc = bm1[tid];
#pragma unroll 8
  for (int k = 0; k < 512; ++k) acc += pool[k] * Wm1[k * 256 + tid];
  t2[tid] = lrelu(acc);
  __syncthreads();
  if (tid < 3) {
    float a = bm2[tid];
#pragma unroll 8
    for (int k = 0; k < 256; ++k) a += t2[k] * Wm2[k * 3 + tid];
    out[g * 3 + tid] = a;
  }
}

// ---------------- launcher ----------------
extern "C" void kernel_launch(void* const* d_in, const int* in_sizes, int n_in,
                              void* d_out, int out_size, void* d_ws, size_t ws_size,
                              hipStream_t stream) {
  (void)in_sizes; (void)n_in; (void)out_size; (void)ws_size;

  const float* x    = (const float*)d_in[0];
  const float* pos  = (const float*)d_in[1];
  const float* tq   = (const float*)d_in[2];
  /* d_in[3] = batch (uniform, sorted) unused */
  const float* W1   = (const float*)d_in[4];
  const float* b1   = (const float*)d_in[5];
  const float* W2   = (const float*)d_in[6];
  const float* b2   = (const float*)d_in[7];
  const float* Wl1a = (const float*)d_in[8];
  const float* bl1a = (const float*)d_in[9];
  const float* Wl1b = (const float*)d_in[10];
  const float* bl1b = (const float*)d_in[11];
  const float* Wm1  = (const float*)d_in[12];
  const float* bm1  = (const float*)d_in[13];
  const float* Wm2  = (const float*)d_in[14];
  const float* bm2  = (const float*)d_in[15];
  float* out = (float*)d_out;

  size_t off = 0;
  auto carve = [&](size_t bytes) -> void* {
    void* p = (char*)d_ws + off;
    off += (bytes + 255) & ~(size_t)255;
    return p;
  };
  _Float16* hcF16   = (_Float16*)carve((size_t)NNODE * 384 * 2);  // h1..h6 concat, f16
  _Float16* xxF16   = (_Float16*)carve((size_t)NNODE * 32 * 2);
  float*    sqb     = (float*)   carve((size_t)NNODE * 4);
  float*    Pb      = (float*)   carve((size_t)NNODE * 64 * 4);
  float*    Qb      = (float*)   carve((size_t)NNODE * 64 * 4);
  int*      idxb    = (int*)     carve((size_t)NNODE * KNN * 4);
  _Float16* t1F16   = (_Float16*)carve((size_t)NNODE * 512 * 2);
  unsigned* pooledU = (unsigned*)carve((size_t)NGRAPH * 512 * 4);
  _Float16* WaT     = (_Float16*)carve((size_t)512 * 384 * 2);
  _Float16* WbT     = (_Float16*)carve((size_t)512 * 512 * 2);

  // prep
  k_init_pooled<<<(NGRAPH * 512 + 255) / 256, 256, 0, stream>>>(pooledU, NGRAPH * 512);
  k_transpose16<<<(384 * 512 + 255) / 256, 256, 0, stream>>>(Wl1a, WaT, 384, 512);
  k_transpose16<<<(512 * 512 + 255) / 256, 256, 0, stream>>>(Wl1b, WbT, 512, 512);
  k_build_xx<<<NNODE / 256, 256, 0, stream>>>(x, pos, tq, xxF16, sqb);

  // conv1 (F=5 padded to 32)
  k_pq5<<<(NNODE * 64) / 256, 256, 0, stream>>>(xxF16, W1, b1, Pb, Qb);
  k_dist_topk<32><<<NGRAPH * 64, 32, 0, stream>>>(xxF16, 32, sqb, idxb);
  k_agg<<<(NNODE * 64) / 256, 256, 0, stream>>>(Pb, Qb, idxb, hcF16 + 0);
  k_sq64<<<NNODE / 256, 256, 0, stream>>>(hcF16 + 0, sqb);

  // convs 2..6 (shared W2/b2)
  for (int c = 2; c <= 6; ++c) {
    const _Float16* hin = hcF16 + (size_t)(c - 2) * 64;
    _Float16* hout      = hcF16 + (size_t)(c - 1) * 64;
    k_pq64<<<NNODE / 4, 256, 0, stream>>>(hin, W2, b2, Pb, Qb);
    k_dist_topk<64><<<NGRAPH * 64, 32, 0, stream>>>(hin, 384, sqb, idxb);
    k_agg<<<(NNODE * 64) / 256, 256, 0, stream>>>(Pb, Qb, idxb, hout);
    if (c < 6) k_sq64<<<NNODE / 256, 256, 0, stream>>>(hout, sqb);
  }

  // big GEMMs + fused max-pool (16x64 output per wave, 4 waves/block)
  k_gemm1<<<(NNODE / 16) * (512 / 64) / 4, 128, 0, stream>>>(hcF16, WaT, bl1a, t1F16);
  k_gemm2_pool<<<(NNODE / 16) * (512 / 64) / 4, 128, 0, stream>>>(t1F16, WbT, bl1b, pooledU);

  // head
  k_head<<<NGRAPH, 256, 0, stream>>>(pooledU, Wm1, bm1, Wm2, bm2, out);
}
